// RPN_36618891165811
// MI455X (gfx1250) — matrix operand order repaired and checked by reading, not compiled, not run on hardware
//
#include <hip/hip_runtime.h>
#include <hip/hip_bf16.h>

typedef __attribute__((ext_vector_type(16))) _Float16 v16h;
typedef __attribute__((ext_vector_type(8)))  _Float16 v8h;
typedef __attribute__((ext_vector_type(8)))  float    v8f;

#define FS    128
#define NPIX  (FS*FS)   // 16384
#define CIN   1024
#define CMID  512
#define BM    64

// Fragment-layout helper (16-bit A 16x32, contiguous k order this file uses for A and B):
// lane = (m%16) + 16*(k/16), halfidx i: k = (lane/16)*16 + i
__device__ __forceinline__ int frag_k(int lane, int i) {
    return ((lane >> 4) << 4) + i;
}

// ---------------------------------------------------------------------------
// Kernel 0a: pack base_w (OIHW fp32) -> f16 A-fragments
// layout: [pos(9)][kcb(32)][ot(32)][lane(32)][i(16)]
// ---------------------------------------------------------------------------
__global__ __launch_bounds__(256)
void pack_w3x3(const float* __restrict__ w, _Float16* __restrict__ wpk) {
    int idx = blockIdx.x * 256 + threadIdx.x;      // < 9*32*32*32*16
    int i    = idx & 15;
    int lane = (idx >> 4) & 31;
    int ot   = (idx >> 9) & 31;
    int kcb  = (idx >> 14) & 31;
    int pos  = idx >> 19;                           // 0..8
    if (pos >= 9) return;
    int oc = ot * 16 + (lane & 15);
    int ic = kcb * 32 + frag_k(lane, i);
    wpk[idx] = (_Float16)w[((size_t)oc * CIN + ic) * 9 + pos];
}

// ---------------------------------------------------------------------------
// Kernel 0b: pack head weights (cls 18 rows || reg 36 rows, padded to 64)
// layout: [kc(16)][mt(4)][lane(32)][i(16)]
// ---------------------------------------------------------------------------
__global__ __launch_bounds__(256)
void pack_heads(const float* __restrict__ cls_w, const float* __restrict__ reg_w,
                _Float16* __restrict__ hpk) {
    int idx = blockIdx.x * 256 + threadIdx.x;      // < 16*4*32*16 = 32768
    if (idx >= 32768) return;
    int i    = idx & 15;
    int lane = (idx >> 4) & 31;
    int mt   = (idx >> 9) & 3;
    int kc   = idx >> 11;                           // 0..15
    int row  = mt * 16 + (lane & 15);
    int k    = kc * 32 + frag_k(lane, i);
    float v = 0.0f;
    if (row < 18)       v = cls_w[row * CMID + k];
    else if (row < 54)  v = reg_w[(row - 18) * CMID + k];
    hpk[idx] = (_Float16)v;
}

// ---------------------------------------------------------------------------
// Kernel 1: 3x3 conv as implicit GEMM, f16 WMMA, fp32 accum, bias+ReLU fused.
// Block: 256 threads (8 waves, 2x4 wave grid), tile 64 oc x 128 pixels (one
// image row). K loop: (kh, kcb) outer with register-held 2x10 input windows,
// all three kw-shifted B tiles staged per sync interval (12 WMMAs between
// barriers). Branch-free staging: float4 interior + clamped halo selects.
// ---------------------------------------------------------------------------
__global__ __launch_bounds__(256)
void conv3x3_wmma(const float* __restrict__ x,
                  const _Float16* __restrict__ wpk,
                  const float* __restrict__ bias,
                  _Float16* __restrict__ ycv) {          // [512][16384] f16
    __shared__ _Float16 bLds[3][8][32][16];              // [kw][ntile][lane][i] 24KB

    const int tid  = threadIdx.x;
    const int lane = tid & 31;
    const int wv   = tid >> 5;
    const int wm   = wv >> 2;        // 0..1 -> 32-row slab
    const int wn   = wv & 3;         // 0..3 -> 32-col slab
    const int h    = blockIdx.y;     // image row, pixels p0..p0+127
    const int p0   = h * FS;

    // staging role: row-pair rp (2 consecutive K rows), col-block cb (8 cols)
    const int kk0 = (tid >> 4) * 2;  // 0,2,..,30
    const int n0  = (tid & 15) * 8;  // 0,8,..,120
    const int lidx = 16 * (kk0 >> 4);     // lane-half offset from K group
    const int i0   = kk0 & 15;            // even -> b32-aligned pair store

    v8f zero = {};
    v8f acc[2][2];
    acc[0][0] = zero; acc[0][1] = zero; acc[1][0] = zero; acc[1][1] = zero;

    for (int kh = 0; kh < 3; ++kh) {
        const int hi = h + kh - 1;
        const bool hok = (hi >= 0) && (hi < FS);   // uniform per block
        for (int kcb = 0; kcb < 32; ++kcb) {
            // ---- register window: 2 rows x 10 cols (cols n0-1 .. n0+8) ----
            _Float16 win[2][10];
            #pragma unroll
            for (int r = 0; r < 2; ++r)
                #pragma unroll
                for (int q = 0; q < 10; ++q) win[r][q] = (_Float16)0.0f;
            if (hok) {
                #pragma unroll
                for (int r = 0; r < 2; ++r) {
                    const float* row = x + (size_t)(kcb * 32 + kk0 + r) * NPIX
                                         + (size_t)hi * FS;
                    float4 a = *(const float4*)(row + n0);
                    float4 b = *(const float4*)(row + n0 + 4);
                    float lft = row[n0 == 0 ? 0 : n0 - 1];
                    float rgt = row[n0 == 120 ? 127 : n0 + 8];
                    lft = (n0 == 0)   ? 0.0f : lft;
                    rgt = (n0 == 120) ? 0.0f : rgt;
                    win[r][0] = (_Float16)lft;
                    win[r][1] = (_Float16)a.x;  win[r][2] = (_Float16)a.y;
                    win[r][3] = (_Float16)a.z;  win[r][4] = (_Float16)a.w;
                    win[r][5] = (_Float16)b.x;  win[r][6] = (_Float16)b.y;
                    win[r][7] = (_Float16)b.z;  win[r][8] = (_Float16)b.w;
                    win[r][9] = (_Float16)rgt;
                    if (kcb < 31) __builtin_prefetch(row + 32 * NPIX, 0, 1);
                }
            }
            // ---- stage all 3 kw-shifted tiles, paired b32 stores ----
            #pragma unroll
            for (int kw = 0; kw < 3; ++kw)
                #pragma unroll
                for (int j = 0; j < 8; ++j) {
                    int n = n0 + j;
                    union { _Float16 hh[2]; unsigned u; } pk;
                    pk.hh[0] = win[0][j + kw];   // row kk0
                    pk.hh[1] = win[1][j + kw];   // row kk0+1
                    *(unsigned*)&bLds[kw][n >> 4][(n & 15) + lidx][i0] = pk.u;
                }
            __syncthreads();

            // ---- 3 positions x (2x2) WMMA between barriers ----
            #pragma unroll
            for (int kw = 0; kw < 3; ++kw) {
                const int pos = kh * 3 + kw;
                v16h afrag[2], bfrag[2];
                #pragma unroll
                for (int mt = 0; mt < 2; ++mt) {
                    int ot = blockIdx.x * 4 + wm * 2 + mt;
                    afrag[mt] = *(const v16h*)(wpk +
                        ((size_t)((pos * 32 + kcb) * 32 + ot)) * 512 + lane * 16);
                }
                #pragma unroll
                for (int nt = 0; nt < 2; ++nt)
                    bfrag[nt] = *(const v16h*)&bLds[kw][wn * 2 + nt][lane][0];
                #pragma unroll
                for (int mt = 0; mt < 2; ++mt)
                    #pragma unroll
                    for (int nt = 0; nt < 2; ++nt)
                        acc[mt][nt] = __builtin_amdgcn_wmma_f32_16x16x32_f16(
                            false, afrag[mt], false, bfrag[nt],
                            (short)0, acc[mt][nt], false, false);
            }
            __syncthreads();
        }
    }

    // ---- epilogue: bias + ReLU, store f16 [oc][pixel] ----
    const int half = lane >> 4;
    const int ncol = lane & 15;
    #pragma unroll
    for (int mt = 0; mt < 2; ++mt)
        #pragma unroll
        for (int nt = 0; nt < 2; ++nt)
            #pragma unroll
            for (int r = 0; r < 8; ++r) {
                int oc = blockIdx.x * BM + wm * 32 + mt * 16 + half * 8 + r;
                int p  = p0 + wn * 32 + nt * 16 + ncol;
                float v = acc[mt][nt][r] + bias[oc];
                v = v > 0.0f ? v : 0.0f;
                ycv[(size_t)oc * NPIX + p] = (_Float16)v;
            }
}

// ---------------------------------------------------------------------------
// Kernel 2: fused heads (1x1 convs via WMMA) + paired softmax + box decode.
// 1 wave per block; block handles 16 pixels. 64x16 GEMM over K=512.
// ---------------------------------------------------------------------------
__global__ __launch_bounds__(32)
void heads_proposals(const _Float16* __restrict__ ycv,
                     const _Float16* __restrict__ hpk,
                     const float* __restrict__ cls_b,
                     const float* __restrict__ reg_b,
                     float* __restrict__ out) {          // [147456][5]
    __shared__ _Float16 bLds[32][16];                    // B tile, frag layout
    __shared__ float    full[64][16];                    // result tile

    const int lane = threadIdx.x;
    const int p0   = blockIdx.x * 16;

    // staging role: 2 consecutive K rows x 8 cols per lane
    const int kk0 = (lane & 15) * 2;    // 0..30
    const int n0  = (lane >> 4) * 8;    // 0 or 8
    const int lidx = 16 * (kk0 >> 4);
    const int i0   = kk0 & 15;

    v8f zero = {};
    v8f acc[4];
    acc[0] = zero; acc[1] = zero; acc[2] = zero; acc[3] = zero;

    for (int kc = 0; kc < 16; ++kc) {
        // stage B: two v8h row segments, paired b32 stores
        v8h r0 = *(const v8h*)(ycv + (size_t)(kc * 32 + kk0) * NPIX + p0 + n0);
        v8h r1 = *(const v8h*)(ycv + (size_t)(kc * 32 + kk0 + 1) * NPIX + p0 + n0);
        #pragma unroll
        for (int j = 0; j < 8; ++j) {
            int n = n0 + j;
            union { _Float16 hh[2]; unsigned u; } pk;
            pk.hh[0] = r0[j];
            pk.hh[1] = r1[j];
            *(unsigned*)&bLds[n + lidx][i0] = pk.u;
        }
        __syncthreads();

        v16h bfrag = *(const v16h*)&bLds[lane][0];
        #pragma unroll
        for (int mt = 0; mt < 4; ++mt) {
            v16h afrag = *(const v16h*)(hpk + ((size_t)(kc * 4 + mt) * 32 + lane) * 16);
            acc[mt] = __builtin_amdgcn_wmma_f32_16x16x32_f16(
                false, afrag, false, bfrag, (short)0, acc[mt], false, false);
        }
        __syncthreads();
    }

    // spill accumulators (+bias) to LDS so each thread can see all 54 channels
    #pragma unroll
    for (int mt = 0; mt < 4; ++mt)
        #pragma unroll
        for (int r = 0; r < 8; ++r) {
            int row = mt * 16 + (lane >> 4) * 8 + r;
            float b = 0.0f;
            if (row < 18)      b = cls_b[row];
            else if (row < 54) b = reg_b[row - 18];
            full[row][lane & 15] = acc[mt][r] + b;
        }
    __syncthreads();

    // anchor wh table: (23,12),(16,16),(11,22) x scales {8,16,32}
    const float AW[9] = {184.f, 368.f, 736.f, 128.f, 256.f, 512.f,  88.f, 176.f, 352.f};
    const float AH[9] = { 96.f, 192.f, 384.f, 128.f, 256.f, 512.f, 176.f, 352.f, 704.f};

    for (int idx = lane; idx < 144; idx += 32) {
        int a = idx >> 4;            // anchor 0..8
        int n = idx & 15;            // pixel within tile
        int p = p0 + n;
        float ax = (float)((p >> 7) << 4);    // h * 16
        float ay = (float)((p & 127) << 4);   // w * 16
        float aw = AW[a], ah = AH[a];
        // softmax pairs channel j with j+9 (reshape (1,2,9F,F)); score = ch 2a+1
        int jc = (2 * a + 1) % 9;
        float l0 = full[jc][n];
        float l1 = full[jc + 9][n];
        float m  = fmaxf(l0, l1);
        float e0 = __expf(l0 - m), e1 = __expf(l1 - m);
        float score = ((2 * a + 1) < 9 ? e0 : e1) / (e0 + e1);
        float r0 = full[18 + 4 * a + 0][n];
        float r1 = full[18 + 4 * a + 1][n];
        float r2 = full[18 + 4 * a + 2][n];
        float r3 = full[18 + 4 * a + 3][n];
        float* o = out + (size_t)(p * 9 + a) * 5;
        o[0] = ax + aw * r0;
        o[1] = ay + ah * r1;
        o[2] = aw + __expf(r2);
        o[3] = ah + __expf(r3);
        o[4] = score;
    }
}

// ---------------------------------------------------------------------------
extern "C" void kernel_launch(void* const* d_in, const int* in_sizes, int n_in,
                              void* d_out, int out_size, void* d_ws, size_t ws_size,
                              hipStream_t stream) {
    const float* x      = (const float*)d_in[0];
    const float* base_w = (const float*)d_in[1];
    const float* base_b = (const float*)d_in[2];
    const float* cls_w  = (const float*)d_in[3];
    const float* cls_b  = (const float*)d_in[4];
    const float* reg_w  = (const float*)d_in[5];
    const float* reg_b  = (const float*)d_in[6];
    float* out = (float*)d_out;

    // workspace layout
    char* ws = (char*)d_ws;
    _Float16* ycv = (_Float16*)ws;                                   // 16 MB
    _Float16* wpk = (_Float16*)(ws + (size_t)CMID * NPIX * 2);       // 9 MB
    _Float16* hpk = (_Float16*)(ws + (size_t)CMID * NPIX * 2
                                   + (size_t)9 * CIN * CMID * 2);    // 64 KB

    // pack weights into WMMA fragment order (f16)
    pack_w3x3<<<(9 * CIN * CMID + 255) / 256, 256, 0, stream>>>(base_w, wpk);
    pack_heads<<<(32768 + 255) / 256, 256, 0, stream>>>(cls_w, reg_w, hpk);

    // main 3x3 conv: grid (512/64 oc-tiles, 128 image rows)
    dim3 g1(CMID / BM, FS);
    conv3x3_wmma<<<g1, 256, 0, stream>>>(x, wpk, base_b, ycv);

    // fused heads + softmax + proposals: one wave per 16 pixels
    heads_proposals<<<NPIX / 16, 32, 0, stream>>>(ycv, hpk, cls_b, reg_b, out);
}
